// ChunkedTropicalAttention_66116726555068
// MI455X (gfx1250) — compile-verified, matrix operand order checked by reading, not run on hardware
//
#include <hip/hip_runtime.h>
#include <math.h>

#define BH 16
#define SEQ 512
#define DK 64
#define DM 512

typedef __attribute__((ext_vector_type(16))) __bf16 v16bf;
typedef __attribute__((ext_vector_type(8)))  __bf16 v8bf;
typedef __attribute__((ext_vector_type(4)))  __bf16 v4bf;
typedef __attribute__((ext_vector_type(8)))  float  v8f;

// Post one lane-wise async global->LDS 128-bit copy (tracked by ASYNCcnt).
__device__ __forceinline__ void async_copy_b128(unsigned lds_byte_off,
                                                const void* gptr) {
  asm volatile("global_load_async_to_lds_b128 %0, %1, off"
               :: "v"(lds_byte_off), "v"((unsigned long long)gptr)
               : "memory");
}
__device__ __forceinline__ void wait_async0() {
  asm volatile("s_wait_asynccnt 0x0" ::: "memory");
}
// Low 32 bits of a generic pointer to LDS == group-relative LDS byte address.
__device__ __forceinline__ unsigned lds_addr32(const void* p) {
  return (unsigned)(unsigned long long)p;
}

// -------------------------------------------------------------------------
// Kernel 1: t = log1p(relu(x)); q/k/v = tropical (max-plus) projections.
// Block = (bh, 32 rows). Weights in LDS with row stride 65 (bank-conflict
// free across lanes that differ in output index).
// -------------------------------------------------------------------------
__global__ void __launch_bounds__(256) tga_proj(
    const float* __restrict__ x,
    const float* __restrict__ Wq, const float* __restrict__ Wk,
    const float* __restrict__ Wv,
    float* __restrict__ q, float* __restrict__ k, float* __restrict__ v)
{
  __shared__ float wqs[64 * 65];
  __shared__ float wks[64 * 65];
  __shared__ float wvs[64 * 65];
  __shared__ float ts[32 * 64];

  const int tid    = threadIdx.x;
  const int bh     = blockIdx.x >> 4;   // 16 row-blocks per bh
  const int rowblk = blockIdx.x & 15;   // 32 rows each
  const int b = bh >> 3, h = bh & 7;

  // Load the three 64x64 weight matrices into padded LDS (16 floats/thread).
#pragma unroll
  for (int c = 0; c < 4; ++c) {
    const int e = tid * 16 + c * 4;
    const int o = e >> 6, i = e & 63;
    const float4 aq = *(const float4*)(Wq + e);
    const float4 ak = *(const float4*)(Wk + e);
    const float4 av = *(const float4*)(Wv + e);
    wqs[o * 65 + i + 0] = aq.x; wqs[o * 65 + i + 1] = aq.y;
    wqs[o * 65 + i + 2] = aq.z; wqs[o * 65 + i + 3] = aq.w;
    wks[o * 65 + i + 0] = ak.x; wks[o * 65 + i + 1] = ak.y;
    wks[o * 65 + i + 2] = ak.z; wks[o * 65 + i + 3] = ak.w;
    wvs[o * 65 + i + 0] = av.x; wvs[o * 65 + i + 1] = av.y;
    wvs[o * 65 + i + 2] = av.z; wvs[o * 65 + i + 3] = av.w;
  }

  // Load 32 rows of x, fuse t = log1p(relu(x)). 8 floats/thread.
#pragma unroll
  for (int c = 0; c < 2; ++c) {
    const int e = tid * 8 + c * 4;
    const int r = e >> 6, i = e & 63;
    const size_t g = ((size_t)(b * SEQ + rowblk * 32 + r)) * DM + h * 64 + i;
    const float4 xv = *(const float4*)(x + g);
    ts[r * 64 + i + 0] = log1pf(fmaxf(xv.x, 0.0f));
    ts[r * 64 + i + 1] = log1pf(fmaxf(xv.y, 0.0f));
    ts[r * 64 + i + 2] = log1pf(fmaxf(xv.z, 0.0f));
    ts[r * 64 + i + 3] = log1pf(fmaxf(xv.w, 0.0f));
  }
  __syncthreads();

  // Max-plus: each thread owns one output column for 8 rows, all 3 mats.
  const int oc = tid & 63;
  const int rg = tid >> 6;
#pragma unroll
  for (int j = 0; j < 8; ++j) {
    const int r = rg * 8 + j;
    float mq = -INFINITY, mk = -INFINITY, mv = -INFINITY;
#pragma unroll 8
    for (int i = 0; i < 64; ++i) {
      const float tv = ts[r * 64 + i];            // LDS broadcast
      mq = fmaxf(mq, tv + wqs[oc * 65 + i]);      // conflict-free (pad 65)
      mk = fmaxf(mk, tv + wks[oc * 65 + i]);
      mv = fmaxf(mv, tv + wvs[oc * 65 + i]);
    }
    const size_t sidx = ((size_t)(bh * SEQ + rowblk * 32 + r)) * DK + oc;
    q[sidx] = mq; k[sidx] = mk; v[sidx] = mv;
  }
}

// -------------------------------------------------------------------------
// Kernel 2: tropical attention. Block = (bh, 32 queries), key tiles of 64.
// Tile fills use the CDNA5 async global->LDS engine (ASYNCcnt) so the VALU
// stays free for the fmax/fmin work and no staging VGPRs are consumed.
// scores[q,k] = -(max_d(qd-kd) - min_d(qd-kd)); ctx = maxplus(scores, v).
// Writes bf16 expm1(ctx) in (b, s, h*64+d) layout -> GEMM A operand.
// -------------------------------------------------------------------------
__global__ void __launch_bounds__(256) tga_attn(
    const float* __restrict__ q, const float* __restrict__ k,
    const float* __restrict__ v, __bf16* __restrict__ abf)
{
  __shared__ float qs[32 * 64];
  __shared__ float kt[64 * 65];
  __shared__ float vt[64 * 65];
  __shared__ float sc[32 * 64];

  const int tid  = threadIdx.x;
  const int bh   = blockIdx.x >> 4;
  const int qblk = blockIdx.x & 15;     // 32 queries per block
  const int b = bh >> 3, h = bh & 7;
  const int s0 = qblk * 32;

  const unsigned qsB = lds_addr32(qs);
  const unsigned ktB = lds_addr32(kt);
  const unsigned vtB = lds_addr32(vt);

  // Async-fill 32x64 query tile (two b128 per thread); completion is
  // covered by the first in-loop s_wait_asynccnt + barrier.
#pragma unroll
  for (int c = 0; c < 2; ++c) {
    const int e = tid * 8 + c * 4;
    async_copy_b128(qsB + (unsigned)e * 4,
                    q + ((size_t)(bh * SEQ + s0 + (e >> 6))) * DK + (e & 63));
  }

  const int oc = tid & 63, qg = tid >> 6;   // ctx ownership: col oc, 8 rows
  const int kc = tid & 63, qh = tid >> 6;   // score ownership: key kc, 8 rows

  float acc[8];
#pragma unroll
  for (int j = 0; j < 8; ++j) acc[j] = -INFINITY;

  for (int kt0 = 0; kt0 < SEQ; kt0 += 64) {
    __syncthreads();   // protect kt/vt/sc from previous iteration

    // Async-fill 64-key k/v tiles: 4 x b128 each per thread, padded rows
    // (stride 65 floats; each 16-float run stays inside one row).
#pragma unroll
    for (int c = 0; c < 4; ++c) {
      const int e = tid * 16 + c * 4;
      const int r = e >> 6, d = e & 63;
      const size_t g = ((size_t)(bh * SEQ + kt0 + r)) * DK + d;
      const unsigned loff = (unsigned)((r * 65 + d) * 4);
      async_copy_b128(ktB + loff, k + g);
      async_copy_b128(vtB + loff, v + g);
    }
    // Prefetch next key tile into GL2 while async engine fills LDS.
    if (kt0 + 64 < SEQ) {
      __builtin_prefetch(k + ((size_t)(bh * SEQ + kt0 + 64)) * DK, 0, 1);
      __builtin_prefetch(v + ((size_t)(bh * SEQ + kt0 + 64)) * DK, 0, 1);
    }
    wait_async0();     // this wave's async fills (incl. q on iter 0) done
    __syncthreads();   // ... and everyone else's

    // Tropical symmetric distance scores: 32q x 64k, 8 pairs/thread.
#pragma unroll
    for (int j = 0; j < 8; ++j) {
      const int qr = qh * 8 + j;
      float mx = -INFINITY, mn = INFINITY;
#pragma unroll 8
      for (int d = 0; d < 64; ++d) {
        const float diff = qs[qr * 64 + d] - kt[kc * 65 + d];  // bcast - conflict-free
        mx = fmaxf(mx, diff);
        mn = fminf(mn, diff);
      }
      sc[qr * 64 + kc] = mn - mx;   // == -(mx - mn)
    }
    __syncthreads();

    // Max-plus context accumulation: 8 rows x 64 keys per thread.
#pragma unroll 4
    for (int kk = 0; kk < 64; ++kk) {
      const float vv = vt[kk * 65 + oc];        // conflict-free (pad 65)
#pragma unroll
      for (int j = 0; j < 8; ++j) {
        acc[j] = fmaxf(acc[j], sc[(qg * 8 + j) * 64 + kk] + vv);  // sc: broadcast
      }
    }
  }

  // Store bf16 expm1(ctx) in GEMM-A layout: A[(b*S+s), h*64 + oc].
#pragma unroll
  for (int j = 0; j < 8; ++j) {
    const int sg = s0 + qg * 8 + j;
    abf[((size_t)(b * SEQ + sg)) * DM + h * 64 + oc] = (__bf16)expm1f(acc[j]);
  }
}

// -------------------------------------------------------------------------
// Kernel 3: convert W_out (512x512 f32) to bf16 once.
// -------------------------------------------------------------------------
__global__ void __launch_bounds__(256) tga_cvt_bf16(
    const float* __restrict__ src, __bf16* __restrict__ dst)
{
  const int i = (blockIdx.x * 256 + threadIdx.x) * 4;
  const float4 f = *(const float4*)(src + i);
  v4bf o;
  o[0] = (__bf16)f.x; o[1] = (__bf16)f.y; o[2] = (__bf16)f.z; o[3] = (__bf16)f.w;
  *(v4bf*)(dst + i) = o;
}

// -------------------------------------------------------------------------
// Kernel 4: out = A @ W^T via v_wmma_f32_16x16x32_bf16, bf16 operands.
// One 16x16 tile per wave32. Per K-step each lane needs exactly two
// contiguous 16B bf16 chunks per fragment -> 4 x global_load_b128 + WMMA.
// All 8 waves of a block share one M-tile row (A reuse through WGP$).
// -------------------------------------------------------------------------
__global__ void __launch_bounds__(256) tga_gemm_wmma(
    const __bf16* __restrict__ A, const __bf16* __restrict__ W,
    float* __restrict__ out)
{
  const int lane = threadIdx.x & 31;
  const int wave = threadIdx.x >> 5;
  const int tile = blockIdx.x * 8 + wave;   // 2048 tiles: 64 M-tiles x 32 N-tiles
  const int mt = tile >> 5;                 // same for all waves in a block
  const int nt = tile & 31;                 // consecutive within a block

  const int mrow  = mt * 16 + (lane & 15);       // A fragment row
  const int nrow  = nt * 16 + (lane & 15);       // B fragment column (W row)
  const int khalf = (lane >= 16) ? 8 : 0;        // 16-bit A/B half-wave K offset

  const __bf16* arow = A + (size_t)mrow * DM;
  const __bf16* wrow = W + (size_t)nrow * DM;

  v8f c = {};
#pragma unroll 4
  for (int k0 = 0; k0 < DM; k0 += 32) {
    const v8bf alo = *(const v8bf*)(arow + k0 + khalf);
    const v8bf ahi = *(const v8bf*)(arow + k0 + 16 + khalf);
    const v8bf blo = *(const v8bf*)(wrow + k0 + khalf);
    const v8bf bhi = *(const v8bf*)(wrow + k0 + 16 + khalf);
    const v16bf af = __builtin_shufflevector(alo, ahi,
        0, 1, 2, 3, 4, 5, 6, 7, 8, 9, 10, 11, 12, 13, 14, 15);
    const v16bf bfr = __builtin_shufflevector(blo, bhi,
        0, 1, 2, 3, 4, 5, 6, 7, 8, 9, 10, 11, 12, 13, 14, 15);
    // (neg_a, A, neg_b, B, c_mod, C, reuse_a, reuse_b)
    c = __builtin_amdgcn_wmma_f32_16x16x32_bf16(false, af, false, bfr,
                                                (short)0, c, false, false);
  }

  // C layout: VGPR r -> M = r (+8 for lanes 16..31), N = lane & 15.
  const int mbase = mt * 16 + ((lane >= 16) ? 8 : 0);
  const int ncol  = nt * 16 + (lane & 15);
#pragma unroll
  for (int r = 0; r < 8; ++r) {
    out[((size_t)(mbase + r)) * DM + ncol] = c[r];
  }
}

// -------------------------------------------------------------------------
extern "C" void kernel_launch(void* const* d_in, const int* in_sizes, int n_in,
                              void* d_out, int out_size, void* d_ws, size_t ws_size,
                              hipStream_t stream) {
  const float* x  = (const float*)d_in[0];
  const float* Wq = (const float*)d_in[1];
  const float* Wk = (const float*)d_in[2];
  const float* Wv = (const float*)d_in[3];
  const float* Wo = (const float*)d_in[4];
  float* out = (float*)d_out;

  float* ws   = (float*)d_ws;
  const size_t QKV = (size_t)BH * SEQ * DK;     // 524288 floats each
  float* qws  = ws;
  float* kws  = ws + QKV;
  float* vws  = ws + 2 * QKV;
  __bf16* abf = (__bf16*)(ws + 3 * QKV);        // 1024 x 512 bf16 (expm1 applied)
  __bf16* wbf = abf + (size_t)(2 * SEQ) * DM;   // 512 x 512 bf16

  (void)in_sizes; (void)n_in; (void)out_size; (void)ws_size;

  tga_cvt_bf16<<<DM * DM / 1024, 256, 0, stream>>>(Wo, wbf);        // independent
  tga_proj<<<BH * 16, 256, 0, stream>>>(x, Wq, Wk, Wv, qws, kws, vws);
  tga_attn<<<BH * 16, 256, 0, stream>>>(qws, kws, vws, abf);
  tga_gemm_wmma<<<256, 256, 0, stream>>>(abf, wbf, out);
}
